// AIMNet2InteractionModule_77223511982115
// MI455X (gfx1250) — matrix operand order repaired and checked by reading, not compiled, not run on hardware
//
#include <hip/hip_runtime.h>
#include <math.h>

// CDNA5 WMMA vector types
typedef __attribute__((ext_vector_type(16))) __bf16 v16bf;
typedef __attribute__((ext_vector_type(8)))  float  v8f;

#define FDIM 128

// ---------------------------------------------------------------------------
// zero helper (grid-stride)
// ---------------------------------------------------------------------------
__global__ void zero_f32_kernel(float* __restrict__ p, long n) {
  long i = (long)blockIdx.x * blockDim.x + threadIdx.x;
  long stride = (long)gridDim.x * blockDim.x;
  for (; i < n; i += stride) p[i] = 0.0f;
}

// ---------------------------------------------------------------------------
// W (f32, row-major [F,F]) -> bf16
// ---------------------------------------------------------------------------
__global__ void cvt_w_bf16_kernel(const float* __restrict__ W,
                                  __bf16* __restrict__ Wb, int n) {
  int i = blockIdx.x * blockDim.x + threadIdx.x;
  if (i < n) Wb[i] = (__bf16)W[i];
}

// ---------------------------------------------------------------------------
// Phase 1: per-pair gather + scatter.
//   weighted[p,f] = fcut[p] * emb[j,f]
//   radial[i,f]  += weighted                      (into out cols F..2F)
//   S[c][i,f]    += u[p,c] * weighted             (c = 0..2)
//   cnt[i]       += 1                             (for bias term)
// 128 threads per pair (one per f), 2 pairs per 256-thread block.
// ---------------------------------------------------------------------------
__global__ void __launch_bounds__(256)
pair_scatter_kernel(const float* __restrict__ emb,
                    const int*   __restrict__ pairlist,
                    const float* __restrict__ fcut,
                    const float* __restrict__ rij,
                    float* __restrict__ out,   // [N, 2F]
                    float* __restrict__ S,     // 3 planes of [N, F]
                    float* __restrict__ cnt,   // [N]
                    int P, int N) {
  long t = (long)blockIdx.x * blockDim.x + threadIdx.x;
  int p = (int)(t >> 7);   // pair index
  int f = (int)(t & 127);  // feature index
  if (p >= P) return;

  int i = pairlist[p];       // idx_i = pairlist[0, p]
  int j = pairlist[P + p];   // idx_j = pairlist[1, p]

  float w = fcut[p] * emb[(long)j * FDIM + f];

  // radial segment-sum straight into output (cols F..2F)
  atomicAdd(&out[(long)i * (2 * FDIM) + FDIM + f], w);

  float rx = rij[3 * p + 0];
  float ry = rij[3 * p + 1];
  float rz = rij[3 * p + 2];
  float inv = rsqrtf(rx * rx + ry * ry + rz * rz);

  long plane = (long)N * FDIM;
  long base  = (long)i * FDIM + f;
  atomicAdd(&S[base],             rx * inv * w);
  atomicAdd(&S[plane + base],     ry * inv * w);
  atomicAdd(&S[2 * plane + base], rz * inv * w);

  if (f == 0) atomicAdd(&cnt[i], 1.0f);
}

// ---------------------------------------------------------------------------
// A-fragment loader: 16x32 bf16 A tile in this file's contiguous k order
// (lane L: row M = L%16, half h = L/16; elems 0..15 <- K = 16h+0..15).
// Converts f32 -> bf16 in registers.
// ---------------------------------------------------------------------------
__device__ __forceinline__ v16bf load_a_frag(const float* __restrict__ arow,
                                             int kb, int half) {
  const float4* c0 = reinterpret_cast<const float4*>(arow + kb + half * 16);
  const float4* c1 = reinterpret_cast<const float4*>(arow + kb + half * 16 + 8);
  float4 p0 = c0[0], p1 = c0[1];
  float4 q0 = c1[0], q1 = c1[1];
  v16bf A;
  A[0]  = (__bf16)p0.x; A[1]  = (__bf16)p0.y; A[2]  = (__bf16)p0.z; A[3]  = (__bf16)p0.w;
  A[4]  = (__bf16)p1.x; A[5]  = (__bf16)p1.y; A[6]  = (__bf16)p1.z; A[7]  = (__bf16)p1.w;
  A[8]  = (__bf16)q0.x; A[9]  = (__bf16)q0.y; A[10] = (__bf16)q0.z; A[11] = (__bf16)q0.w;
  A[12] = (__bf16)q1.x; A[13] = (__bf16)q1.y; A[14] = (__bf16)q1.z; A[15] = (__bf16)q1.w;
  return A;
}

// One GEMM row-strip: acc(16x16) = S_plane_tile(16x128) @ B(128x16), K-split
// into four 16x16x32 WMMAs reusing the pre-loaded B fragments.
__device__ __forceinline__ v8f gemm_plane(const float* __restrict__ arow,
                                          int half,
                                          v16bf B0, v16bf B1, v16bf B2, v16bf B3) {
  v8f acc = {};
  acc = __builtin_amdgcn_wmma_f32_16x16x32_bf16(false, load_a_frag(arow,  0, half),
                                                false, B0, (short)0, acc, false, false);
  acc = __builtin_amdgcn_wmma_f32_16x16x32_bf16(false, load_a_frag(arow, 32, half),
                                                false, B1, (short)0, acc, false, false);
  acc = __builtin_amdgcn_wmma_f32_16x16x32_bf16(false, load_a_frag(arow, 64, half),
                                                false, B2, (short)0, acc, false, false);
  acc = __builtin_amdgcn_wmma_f32_16x16x32_bf16(false, load_a_frag(arow, 96, half),
                                                false, B3, (short)0, acc, false, false);
  return acc;
}

// ---------------------------------------------------------------------------
// Phase 2: per 16-atom tile, compute vec_c = S_c @ W^T + cnt*b with WMMA,
// then out[i,g] = sqrt(sum_c vec_c^2 + 1e-12) into cols 0..F of out.
//
// Block = 256 threads = 8 waves; wave w owns output columns [16w, 16w+16).
// Each wave: loads the 4 K-tiles of B once (v16bf x4), then runs 3 planes
// x 4 k-steps = 12 v_wmma_f32_16x16x32_bf16, all accumulators in VGPRs
// (three named v8f accs -- no runtime-indexed array, no scratch spills).
//
// Fragment packing used here (wave32; A and B share contiguous k order):
//  B (32x16 bf16): lane L, col N=L%16, half h:
//     elems 0..15 <- K = 16h + 0..15   (B[k][g] = W[g][k], contiguous in k)
//  D (16x16 f32): VGPR r, lane L -> row M = r + 8h, col N = L%16
// ---------------------------------------------------------------------------
__global__ void __launch_bounds__(256)
vec_gemm_norm_kernel(const float*  __restrict__ S,    // 3 planes [N,F] f32
                     const __bf16* __restrict__ Wb,   // [F,F] bf16 row-major
                     const float*  __restrict__ bias, // [F]
                     const float*  __restrict__ cnt,  // [N]
                     float* __restrict__ out,         // [N, 2F]
                     int N) {
  const int lane = threadIdx.x & 31;
  const int wave = threadIdx.x >> 5;   // 0..7
  const int row  = lane & 15;
  const int half = lane >> 4;
  const int g_base    = wave * 16;
  const int atom_base = blockIdx.x * 16;
  const long plane = (long)N * FDIM;

  // A-row index (clamped for ragged last tile; N=20000 is a multiple of 16)
  int ar = atom_base + row;
  if (ar >= N) ar = N - 1;
  const float* arow0 = S + (long)ar * FDIM;            // plane 0
  const float* arow1 = arow0 + plane;                  // plane 1
  const float* arow2 = arow1 + plane;                  // plane 2

  // B fragments: four K-tiles, each 16 contiguous bf16 from W row g
  const __bf16* brow = Wb + (long)(g_base + row) * FDIM + half * 16;
  v16bf B0 = *reinterpret_cast<const v16bf*>(brow);
  v16bf B1 = *reinterpret_cast<const v16bf*>(brow + 32);
  v16bf B2 = *reinterpret_cast<const v16bf*>(brow + 64);
  v16bf B3 = *reinterpret_cast<const v16bf*>(brow + 96);

  // Three spatial planes -> three register-resident accumulators
  v8f acc0 = gemm_plane(arow0, half, B0, B1, B2, B3);
  v8f acc1 = gemm_plane(arow1, half, B0, B1, B2, B3);
  v8f acc2 = gemm_plane(arow2, half, B0, B1, B2, B3);

  // Norm over the 3 spatial components + bias term, store cols 0..F.
  const int g  = g_base + row;
  const float bg = bias[g];
#pragma unroll
  for (int r = 0; r < 8; ++r) {
    int am = atom_base + r + half * 8;
    if (am < N) {
      float cv = cnt[am];
      float v0 = acc0[r] + cv * bg;
      float v1 = acc1[r] + cv * bg;
      float v2 = acc2[r] + cv * bg;
      out[(long)am * (2 * FDIM) + g] = sqrtf(v0 * v0 + v1 * v1 + v2 * v2 + 1e-12f);
    }
  }
}

// ---------------------------------------------------------------------------
// Launcher
// ---------------------------------------------------------------------------
extern "C" void kernel_launch(void* const* d_in, const int* in_sizes, int n_in,
                              void* d_out, int out_size, void* d_ws, size_t ws_size,
                              hipStream_t stream) {
  const float* emb  = (const float*)d_in[0];  // [N, F]
  const int*   pl   = (const int*)  d_in[1];  // [2, P]
  const float* fcut = (const float*)d_in[2];  // [P, 1]
  const float* rij  = (const float*)d_in[3];  // [P, 3]
  const float* W    = (const float*)d_in[4];  // [F, F]
  const float* bias = (const float*)d_in[5];  // [F]
  float* out = (float*)d_out;                 // [N, 2F]

  const int N = in_sizes[0] / FDIM;
  const int P = in_sizes[1] / 2;
  const long plane = (long)N * FDIM;

  // workspace layout: S (3 planes f32) | cnt (f32) | Wb (bf16, 64B aligned)
  float* S   = (float*)d_ws;
  float* cnt = S + 3 * plane;
  size_t wb_off = (((size_t)(3 * plane + N) * sizeof(float)) + 63) & ~(size_t)63;
  __bf16* Wb = (__bf16*)((char*)d_ws + wb_off);

  // Phase 0: clear atomic targets
  zero_f32_kernel<<<1024, 256, 0, stream>>>(S, 3 * plane + N);
  zero_f32_kernel<<<1024, 256, 0, stream>>>(out, (long)out_size);

  // Phase 0b: W -> bf16
  cvt_w_bf16_kernel<<<(FDIM * FDIM + 255) / 256, 256, 0, stream>>>(W, Wb, FDIM * FDIM);

  // Phase 1: pair gather/scatter (128 threads per pair)
  long pthreads = (long)P * 128;
  int pblocks = (int)((pthreads + 255) / 256);
  pair_scatter_kernel<<<pblocks, 256, 0, stream>>>(emb, pl, fcut, rij, out, S, cnt, P, N);

  // Phase 2: WMMA GEMM + norm fuse
  int atiles = (N + 15) / 16;
  vec_gemm_norm_kernel<<<atiles, 256, 0, stream>>>(S, Wb, bias, cnt, out, N);
}
